// EGNN_12128987644271
// MI455X (gfx1250) — matrix operand order repaired and checked by reading, not compiled, not run on hardware
//
#include <hip/hip_runtime.h>

typedef __bf16 bf16;
typedef __attribute__((ext_vector_type(8)))  float v8f;
typedef __attribute__((ext_vector_type(16))) bf16  v16bf;
typedef __attribute__((ext_vector_type(8)))  bf16  v8bf;

#define IN_DIM   512
#define EDGE_DIM 51
#define EK   1088   // edge-MLP input K (1076 padded to 32)
#define EKS  1104   // LDS row stride for e_in
#define NK   1024   // node-MLP input K
#define NKS  1040
#define AS   528    // 512-wide activation LDS stride
#define MBLK 64     // rows (edges/nodes) per workgroup  -> 4 M-tiles
#define MT4  4      // MBLK/16

// ---------------------------------------------------------------------------
// WMMA fragment loaders.
// A (16x32 bf16, per ISA table): lane L holds row M=L%16; lanes 0-15 carry
// K = kt*32 + {0..7, 16..23}, lanes 16-31 carry K = kt*32 + {8..15, 24..31}.
// B (32x16 bf16) is PRE-PACKED per-lane in global memory: fragment `frag`
// occupies 512 contiguous halves; lane L reads 16 contiguous halves.
// ---------------------------------------------------------------------------
__device__ __forceinline__ v16bf ldA(const bf16* s, int stride, int kt, int lane) {
  const bf16* p = s + (lane & 15) * stride + kt * 32 + ((lane >> 4) << 3);
  v16bf a;
  ((v8bf*)&a)[0] = *(const v8bf*)p;        // ds_load_b128
  ((v8bf*)&a)[1] = *(const v8bf*)(p + 16); // ds_load_b128
  return a;
}

__device__ __forceinline__ v16bf ldB(const bf16* Wp, size_t frag, int lane) {
  return *(const v16bf*)(Wp + (frag * 32 + (size_t)lane) * 16); // 2x global_load_b128
}

__device__ __forceinline__ float silu_f(float x) {
  return x / (1.f + __expf(-x));
}

// Register-blocked tile GEMM: out[MT*16 x 8*NTW*16] = act(in @ Wp^T + bias).
// 8 wave32 per block; each wave owns NTW N-tiles across all MT M-tiles, so a
// B fragment loaded once from L2 feeds MT WMMAs (4x weight-traffic reduction).
template<int MT, int NTW, bool SILU>
__device__ __forceinline__ void gemm_tile(const bf16* sIn, int inStride, int numKt,
                                          const bf16* Wp, const float* bias,
                                          bf16* sOut, int outStride)
{
  const int tid  = threadIdx.x;
  const int wave = tid >> 5, lane = tid & 31;
  v8f acc[MT][NTW] = {};
  for (int kt = 0; kt < numKt; ++kt) {
    v16bf a[MT];
#pragma unroll
    for (int mt = 0; mt < MT; ++mt)
      a[mt] = ldA(sIn + mt * 16 * inStride, inStride, kt, lane);
#pragma unroll
    for (int t = 0; t < NTW; ++t) {
      v16bf b = ldB(Wp, (size_t)(wave * NTW + t) * numKt + kt, lane);
#pragma unroll
      for (int mt = 0; mt < MT; ++mt)
        acc[mt][t] = __builtin_amdgcn_wmma_f32_16x16x32_bf16(false, a[mt], false, b,
                                                             (short)0, acc[mt][t], false, false);
    }
  }
#pragma unroll
  for (int t = 0; t < NTW; ++t) {
    const int   n  = (wave * NTW + t) * 16 + (lane & 15);
    const float bn = bias[n];
    const int   mb = (lane >> 4) << 3;  // C layout: VGPR v -> M = v + 8*(lane/16)
#pragma unroll
    for (int mt = 0; mt < MT; ++mt) {
#pragma unroll
      for (int v = 0; v < 8; ++v) {
        float x = acc[mt][t][v] + bn;
        if (SILU) x = silu_f(x);
        sOut[(mt * 16 + mb + v) * outStride + n] = (bf16)x;
      }
    }
  }
}

// ---------------------------------------------------------------------------
// Prep kernels
// ---------------------------------------------------------------------------
__global__ void init_kernel(float* agg, float* posOut, const float* pos,
                            int nAgg, int nPos) {
  int i = blockIdx.x * blockDim.x + threadIdx.x;
  if (i < nAgg) agg[i] = 0.f;
  if (i < nPos) posOut[i] = pos[i];
}

__global__ void cvt_bf16_kernel(const float* __restrict__ src, bf16* __restrict__ dst, int n) {
  int i = blockIdx.x * blockDim.x + threadIdx.x;
  if (i < n) dst[i] = (bf16)src[i];
}

// Pack W[outD][inD] (f32, row-major; y = x @ W^T) into per-lane WMMA B frags:
// dst[((nt*numKt+kt)*32 + lane)*16 + j] = W[nt*16 + lane%16][kt*32 + j%8 + 8*(lane/16) + 16*(j/8)]
__global__ void pack_weight_kernel(const float* __restrict__ W, bf16* __restrict__ dst,
                                   int inD, int numKt, int total) {
  int p = blockIdx.x * blockDim.x + threadIdx.x;
  if (p >= total) return;
  int j    = p & 15;
  int lane = (p >> 4) & 31;
  int frag = p >> 9;
  int kt   = frag % numKt;
  int nt   = frag / numKt;
  int N = nt * 16 + (lane & 15);
  int K = kt * 32 + (j & 7) + ((lane >> 4) << 3) + ((j >> 3) << 4);
  float v = (K < inD) ? W[(size_t)N * inD + K] : 0.f;
  dst[p] = (bf16)v;
}

// ---------------------------------------------------------------------------
// Edge kernel: 1 block = 64 edges. Full edge MLP + coord MLP + scatters.
// LDS: e_in 64x1104 bf16 (141KB, aliased as 64x528 ping buffer after L0)
//      + act 64x528 bf16 (67.6KB) + geometry (~1.5KB) = ~210KB (<320KB/WGP).
// ---------------------------------------------------------------------------
__global__ __launch_bounds__(256) void egnn_edge_kernel(
    const bf16* __restrict__ hbf, const long long* __restrict__ ei,
    const float* __restrict__ ea, const float* __restrict__ pos,
    const bf16* __restrict__ ew0p, const float* __restrict__ eb0,
    const bf16* __restrict__ ew1p, const float* __restrict__ eb1,
    const bf16* __restrict__ ew2p, const float* __restrict__ eb2,
    const bf16* __restrict__ cw0p, const float* __restrict__ cb0,
    const bf16* __restrict__ cw1p, const float* __restrict__ cb1,
    const float* __restrict__ cw2,
    float* agg, float* posOut, int E)
{
  extern __shared__ char smem[];
  bf16*  sIn  = (bf16*)smem;                  // MBLK x EKS (aliased as sB after L0)
  bf16*  sA1  = sIn + MBLK * EKS;             // MBLK x AS
  float* sCD  = (float*)(sA1 + MBLK * AS);    // MBLK x 3 coord_diff
  float* sD2  = sCD + MBLK * 3;               // MBLK dist2
  int*   sRow = (int*)(sD2 + MBLK);           // MBLK
  int*   sCol = sRow + MBLK;                  // MBLK
  bf16*  sB   = sIn;                          // alias (sIn dead after layer 0)

  const int tid = threadIdx.x;
  const int tileBase = blockIdx.x * MBLK;

  // per-edge geometry
  if (tid < MBLK) {
    int e = tileBase + tid;
    long long r = 0, c = 0;
    float d2 = 1e-8f, cdx = 0.f, cdy = 0.f, cdz = 0.f;
    if (e < E) {
      r = ei[e]; c = ei[(size_t)E + e];
      float dx = pos[r*3+0] - pos[c*3+0];
      float dy = pos[r*3+1] - pos[c*3+1];
      float dz = pos[r*3+2] - pos[c*3+2];
      float s  = dx*dx + dy*dy + dz*dz;
      d2 = fminf(fmaxf(s, 1e-8f), 100.f);
      float nrm = sqrtf(d2) + 1e-8f;
      cdx = dx / nrm; cdy = dy / nrm; cdz = dz / nrm;
    }
    sRow[tid] = (int)r; sCol[tid] = (int)c; sD2[tid] = d2;
    sCD[tid*3+0] = cdx; sCD[tid*3+1] = cdy; sCD[tid*3+2] = cdz;
  }
  __syncthreads();

  // stage e_in = [h[row] | h[col] | edge_attr | dist2 | 0-pad] as bf16
  {
    int m = tid >> 2, ch = tid & 3;       // 4 threads per row
    bool valid = (tileBase + m) < E;
    size_t rbase = (size_t)sRow[m] * IN_DIM;
    size_t cbase = (size_t)sCol[m] * IN_DIM;
    float  d2    = sD2[m];
    for (int k0 = ch * 8; k0 < EK; k0 += 32) {
      v8bf v;
      if (valid && k0 + 8 <= IN_DIM)
        v = *(const v8bf*)(hbf + rbase + k0);
      else if (valid && k0 >= IN_DIM && k0 + 8 <= 2*IN_DIM)
        v = *(const v8bf*)(hbf + cbase + (k0 - IN_DIM));
      else {
#pragma unroll
        for (int j = 0; j < 8; ++j) {
          int k = k0 + j; float x = 0.f;
          if (valid) {
            if (k < 2*IN_DIM + EDGE_DIM)       x = ea[(size_t)(tileBase + m)*EDGE_DIM + (k - 2*IN_DIM)];
            else if (k == 2*IN_DIM + EDGE_DIM) x = d2;
          }
          ((bf16*)&v)[j] = (bf16)x;
        }
      }
      *(v8bf*)(sIn + m * EKS + k0) = v;
    }
  }
  __syncthreads();

  gemm_tile<MT4, 4, true>(sIn, EKS, EK/32, ew0p, eb0, sA1, AS);  // 1088 -> 512, SiLU
  __syncthreads();
  gemm_tile<MT4, 4, true>(sA1, AS, 16, ew1p, eb1, sB, AS);       // 512 -> 512, SiLU
  __syncthreads();

  // layer 2: 512 -> 512 linear; scatter-add e into agg_node[row]; keep bf16 copy
  {
    const int wave = tid >> 5, lane = tid & 31;
    v8f acc[MT4][4] = {};
    for (int kt = 0; kt < 16; ++kt) {
      v16bf a[MT4];
#pragma unroll
      for (int mt = 0; mt < MT4; ++mt)
        a[mt] = ldA(sB + mt * 16 * AS, AS, kt, lane);
#pragma unroll
      for (int t = 0; t < 4; ++t) {
        v16bf b = ldB(ew2p, (size_t)(wave*4 + t)*16 + kt, lane);
#pragma unroll
        for (int mt = 0; mt < MT4; ++mt)
          acc[mt][t] = __builtin_amdgcn_wmma_f32_16x16x32_bf16(false, a[mt], false, b,
                                                               (short)0, acc[mt][t], false, false);
      }
    }
#pragma unroll
    for (int t = 0; t < 4; ++t) {
      const int   n  = (wave*4 + t)*16 + (lane & 15);
      const float bn = eb2[n];
      const int   mb = (lane >> 4) << 3;
#pragma unroll
      for (int mt = 0; mt < MT4; ++mt) {
#pragma unroll
        for (int v = 0; v < 8; ++v) {
          int m = mt * 16 + mb + v;
          float x = acc[mt][t][v] + bn;
          if (tileBase + m < E)
            atomicAdd(agg + (size_t)sRow[m] * IN_DIM + n, x);  // global_atomic_add_f32
          sA1[m * AS + n] = (bf16)x;
        }
      }
    }
  }
  __syncthreads();

  gemm_tile<MT4, 2, true>(sA1, AS, 16, cw0p, cb0, sB, AS);   // 512 -> 256, SiLU
  __syncthreads();
  gemm_tile<MT4, 2, true>(sB, AS, 8, cw1p, cb1, sA1, AS);    // 256 -> 256, SiLU
  __syncthreads();

  // coord head: 256 -> 1, clip, trans = coord_diff * cu, scatter into pos_out
  if (tid < MBLK && (tileBase + tid) < E) {
    float s = 0.f;
    for (int k = 0; k < 256; ++k) s += (float)sA1[tid * AS + k] * cw2[k];
    s = fminf(fmaxf(s, -1.f), 1.f);
#pragma unroll
    for (int d = 0; d < 3; ++d)
      atomicAdd(posOut + (size_t)sRow[tid] * 3 + d, sCD[tid*3 + d] * s);
  }
}

// ---------------------------------------------------------------------------
// Node kernel: 1 block = 64 nodes. x_in = [h | agg] -> MLP -> residual out.
// ---------------------------------------------------------------------------
__global__ __launch_bounds__(256) void egnn_node_kernel(
    const float* __restrict__ h, const bf16* __restrict__ hbf,
    const float* __restrict__ agg,
    const bf16* __restrict__ nw0p, const float* __restrict__ nb0,
    const bf16* __restrict__ nw1p, const float* __restrict__ nb1,
    const bf16* __restrict__ nw2p, const float* __restrict__ nb2,
    float* hOut, int N)
{
  extern __shared__ char smem[];
  bf16* sIn = (bf16*)smem;         // MBLK x NKS (aliased as sB after layer 0)
  bf16* sA1 = sIn + MBLK * NKS;    // MBLK x AS
  bf16* sB  = sIn;
  const int tid  = threadIdx.x;
  const int base = blockIdx.x * MBLK;

  {
    int m = tid >> 2, ch = tid & 3;
    int node = base + m;
    bool valid = node < N;
    for (int k0 = ch * 8; k0 < NK; k0 += 32) {
      v8bf v;
      if (valid && k0 + 8 <= IN_DIM)
        v = *(const v8bf*)(hbf + (size_t)node * IN_DIM + k0);
      else {
#pragma unroll
        for (int j = 0; j < 8; ++j) {
          float x = valid ? agg[(size_t)node * IN_DIM + (k0 - IN_DIM) + j] : 0.f;
          ((bf16*)&v)[j] = (bf16)x;
        }
      }
      *(v8bf*)(sIn + m * NKS + k0) = v;
    }
  }
  __syncthreads();

  gemm_tile<MT4, 4, true>(sIn, NKS, NK/32, nw0p, nb0, sA1, AS);  // 1024 -> 512, SiLU
  __syncthreads();
  gemm_tile<MT4, 4, true>(sA1, AS, 16, nw1p, nb1, sB, AS);       // 512 -> 512, SiLU
  __syncthreads();

  {  // 512 -> 512 linear + residual, f32 out
    const int wave = tid >> 5, lane = tid & 31;
    v8f acc[MT4][4] = {};
    for (int kt = 0; kt < 16; ++kt) {
      v16bf a[MT4];
#pragma unroll
      for (int mt = 0; mt < MT4; ++mt)
        a[mt] = ldA(sB + mt * 16 * AS, AS, kt, lane);
#pragma unroll
      for (int t = 0; t < 4; ++t) {
        v16bf b = ldB(nw2p, (size_t)(wave*4 + t)*16 + kt, lane);
#pragma unroll
        for (int mt = 0; mt < MT4; ++mt)
          acc[mt][t] = __builtin_amdgcn_wmma_f32_16x16x32_bf16(false, a[mt], false, b,
                                                               (short)0, acc[mt][t], false, false);
      }
    }
#pragma unroll
    for (int t = 0; t < 4; ++t) {
      const int   n  = (wave*4 + t)*16 + (lane & 15);
      const float bn = nb2[n];
      const int   mb = (lane >> 4) << 3;
#pragma unroll
      for (int mt = 0; mt < MT4; ++mt) {
#pragma unroll
        for (int v = 0; v < 8; ++v) {
          int node = base + mt * 16 + mb + v;
          if (node < N) {
            size_t idx = (size_t)node * IN_DIM + n;
            hOut[idx] = h[idx] + acc[mt][t][v] + bn;
          }
        }
      }
    }
  }
}

// ---------------------------------------------------------------------------
extern "C" void kernel_launch(void* const* d_in, const int* in_sizes, int n_in,
                              void* d_out, int out_size, void* d_ws, size_t ws_size,
                              hipStream_t stream)
{
  const float*     h   = (const float*)d_in[0];
  const long long* ei  = (const long long*)d_in[1];  // int64 edge_index [2,E]
  const float*     ea  = (const float*)d_in[2];
  const float*     pos = (const float*)d_in[3];
  const float* ew0 = (const float*)d_in[4];  const float* eb0 = (const float*)d_in[5];
  const float* ew1 = (const float*)d_in[6];  const float* eb1 = (const float*)d_in[7];
  const float* ew2 = (const float*)d_in[8];  const float* eb2 = (const float*)d_in[9];
  const float* cw0 = (const float*)d_in[10]; const float* cb0 = (const float*)d_in[11];
  const float* cw1 = (const float*)d_in[12]; const float* cb1 = (const float*)d_in[13];
  const float* cw2 = (const float*)d_in[14];
  const float* nw0 = (const float*)d_in[15]; const float* nb0 = (const float*)d_in[16];
  const float* nw1 = (const float*)d_in[17]; const float* nb1 = (const float*)d_in[18];
  const float* nw2 = (const float*)d_in[19]; const float* nb2 = (const float*)d_in[20];

  const int N = in_sizes[0] / IN_DIM;
  const int E = in_sizes[1] / 2;

  float* hOut   = (float*)d_out;
  float* posOut = hOut + (size_t)N * IN_DIM;

  // workspace carve (~36 MB): agg[N,512] f32, h in bf16, packed bf16 weights
  char* ws = (char*)d_ws;
  size_t off = 0;
  auto take = [&](size_t bytes) -> char* {
    char* p = ws + off;
    off += (bytes + 255) & ~(size_t)255;
    return p;
  };
  float* agg  = (float*)take((size_t)N * IN_DIM * sizeof(float));
  bf16*  hbf  = (bf16*) take((size_t)N * IN_DIM * sizeof(bf16));
  bf16*  ew0p = (bf16*) take((size_t)32 * (EK/32) * 512 * 2);
  bf16*  ew1p = (bf16*) take((size_t)32 * 16 * 512 * 2);
  bf16*  ew2p = (bf16*) take((size_t)32 * 16 * 512 * 2);
  bf16*  cw0p = (bf16*) take((size_t)16 * 16 * 512 * 2);
  bf16*  cw1p = (bf16*) take((size_t)16 *  8 * 512 * 2);
  bf16*  nw0p = (bf16*) take((size_t)32 * 32 * 512 * 2);
  bf16*  nw1p = (bf16*) take((size_t)32 * 16 * 512 * 2);
  bf16*  nw2p = (bf16*) take((size_t)32 * 16 * 512 * 2);

  const size_t shEdge = (size_t)(MBLK*EKS + MBLK*AS) * sizeof(bf16)
                      + (size_t)MBLK * 4 * sizeof(float) + (size_t)MBLK * 2 * sizeof(int);
  const size_t shNode = (size_t)(MBLK*NKS + MBLK*AS) * sizeof(bf16);
  // >64KB dynamic LDS: opt in (no-op if already set; not a stream operation)
  (void)hipFuncSetAttribute((const void*)egnn_edge_kernel,
                            hipFuncAttributeMaxDynamicSharedMemorySize, (int)shEdge);
  (void)hipFuncSetAttribute((const void*)egnn_node_kernel,
                            hipFuncAttributeMaxDynamicSharedMemorySize, (int)shNode);

  // 1) zero agg, pos -> posOut
  {
    int nAgg = N * IN_DIM, nPos = N * 3;
    init_kernel<<<(nAgg + 255) / 256, 256, 0, stream>>>(agg, posOut, pos, nAgg, nPos);
  }
  // 2) h -> bf16
  {
    int n = N * IN_DIM;
    cvt_bf16_kernel<<<(n + 255) / 256, 256, 0, stream>>>(h, hbf, n);
  }
  // 3) pack all weights into WMMA B-fragment layout (bf16)
  auto pack = [&](const float* W, bf16* dst, int inD, int numKt, int outD) {
    int total = (outD / 16) * numKt * 512;
    pack_weight_kernel<<<(total + 255) / 256, 256, 0, stream>>>(W, dst, inD, numKt, total);
  };
  pack(ew0, ew0p, 2*IN_DIM + EDGE_DIM + 1, EK/32, 512);
  pack(ew1, ew1p, 512, 16, 512);
  pack(ew2, ew2p, 512, 16, 512);
  pack(cw0, cw0p, 512, 16, 256);
  pack(cw1, cw1p, 256,  8, 256);
  pack(nw0, nw0p, 1024, 32, 512);
  pack(nw1, nw1p, 512, 16, 512);
  pack(nw2, nw2p, 512, 16, 512);

  // 4) edge pipeline (64 edges / block)
  {
    int blocks = (E + MBLK - 1) / MBLK;
    egnn_edge_kernel<<<blocks, 256, shEdge, stream>>>(
        hbf, ei, ea, pos,
        ew0p, eb0, ew1p, eb1, ew2p, eb2,
        cw0p, cb0, cw1p, cb1, cw2,
        agg, posOut, E);
  }
  // 5) node pipeline (64 nodes / block)
  {
    int blocks = (N + MBLK - 1) / MBLK;
    egnn_node_kernel<<<blocks, 256, shNode, stream>>>(
        h, hbf, agg, nw0p, nb0, nw1p, nb1, nw2p, nb2, hOut, N);
  }
  (void)n_in; (void)out_size; (void)ws_size;
}
